// HausdorffDTLoss_85229331022122
// MI455X (gfx1250) — matrix operand order repaired and checked by reading, not compiled, not run on hardware
//
#include <hip/hip_runtime.h>
#include <math.h>

// ---------------------------------------------------------------------------
// HausdorffDTLoss for MI455X (gfx1250, wave32).
//   K1: EDT distance-field of target masks            (R blocks)
//   K2: EDT distance-field of sigmoid(pred) masks,
//       dist[r,pos] = pdt^2 + (c==label ? tdt^2 : 0)  (R*C blocks)
//   K3: cumdist = TRIL(ones)[256x256] @ dist[256,P] via v_wmma_f32_16x16x32_f16
//       (triangular chunks skipped with scalar branches; all-ones chunks use a
//        constant A tile), fused with err = (sigmoid(x)-tr)^2 dot + reduction.
//       dist chunks staged into LDS with GLOBAL_LOAD_ASYNC_TO_LDS_B32
//       (ASYNCcnt path) — builtin confirmed present on this toolchain.
//   K4: deterministic tree reduction of per-block partials -> scalar
// ---------------------------------------------------------------------------

typedef __attribute__((ext_vector_type(16))) _Float16 v16h;
typedef __attribute__((ext_vector_type(8)))  float    v8f;

#define R_   256
#define C_   80
#define W_   28
#define HW_  784            // 28*28
#define P_   (C_ * HW_)     // 62720 positions per r
#define NT_  (P_ / 16)      // 3920 n-tiles
#define INF_ 1e10f

#if defined(__gfx1250__) && __has_builtin(__builtin_amdgcn_global_load_async_to_lds_b32) && __has_builtin(__builtin_amdgcn_s_wait_asynccnt)
#define USE_ASYNC_LDS 1
#else
#define USE_ASYNC_LDS 0
#endif

#if USE_ASYNC_LDS
typedef __attribute__((address_space(1))) int* gint_p;
typedef __attribute__((address_space(3))) int* lint_p;
__device__ __forceinline__ void async_copy_b32(const float* g, float* l) {
  // generic->AS1 / generic->AS3 via integer casts (LDS generic addr low 32 bits
  // are the LDS offset per the aperture encoding). Builtin wants int* params.
  gint_p gp = (gint_p)(uintptr_t)g;
  lint_p lp = (lint_p)(unsigned)(uintptr_t)l;
  __builtin_amdgcn_global_load_async_to_lds_b32(gp, lp, 0, 0);
}
#endif

// ---------------- shared helpers (EDT) -------------------------------------

__device__ __forceinline__ float block_max(float v, float* red, int tid) {
  red[tid] = v;
  __syncthreads();
  for (int s = 128; s > 0; s >>= 1) {
    if (tid < s) red[tid] = fmaxf(red[tid], red[tid + s]);
    __syncthreads();
  }
  float m = red[0];
  __syncthreads();
  return m;
}

// exact 1D min-plus envelope along W then H; writes sqrt(d) back into a[]
__device__ __forceinline__ void minplus_2d(float* a, float* b, int tid) {
  for (int pix = tid; pix < HW_; pix += 256) {
    int h = pix / W_, i = pix % W_;
    const float* row = a + h * W_;
    float best = INF_;
    #pragma unroll
    for (int j = 0; j < W_; ++j) {
      float d = (float)(i - j);
      best = fminf(best, row[j] + d * d);
    }
    b[pix] = best;
  }
  __syncthreads();
  for (int pix = tid; pix < HW_; pix += 256) {
    int h = pix / W_, w = pix % W_;
    float best = INF_;
    #pragma unroll
    for (int j = 0; j < W_; ++j) {
      float d = (float)(h - j);
      best = fminf(best, b[j * W_ + w] + d * d);
    }
    a[pix] = sqrtf(best);
  }
  __syncthreads();
}

// msk[pix] in {0,1}. On return, fgd[pix] holds the normalized distance field.
__device__ __forceinline__ void distance_field(float* msk, float* a, float* b,
                                               float* fgd, float* red, int tid) {
  for (int pix = tid; pix < HW_; pix += 256) a[pix] = (msk[pix] > 0.5f) ? INF_ : 0.f;
  __syncthreads();
  minplus_2d(a, b, tid);
  float lmax = 0.f, lany = 0.f;
  for (int pix = tid; pix < HW_; pix += 256) {
    fgd[pix] = a[pix];
    lmax = fmaxf(lmax, a[pix]);
    lany = fmaxf(lany, msk[pix]);
  }
  float fg_max = block_max(lmax, red, tid);
  float fg_any = block_max(lany, red, tid);

  for (int pix = tid; pix < HW_; pix += 256) a[pix] = (msk[pix] > 0.5f) ? 0.f : INF_;
  __syncthreads();
  minplus_2d(a, b, tid);
  float lbmax = 0.f, lbany = 0.f;
  for (int pix = tid; pix < HW_; pix += 256) {
    lbmax = fmaxf(lbmax, a[pix]);
    lbany = fmaxf(lbany, 1.f - msk[pix]);
  }
  float bg_max = block_max(lbmax, red, tid);
  float bg_any = block_max(lbany, red, tid);

  fg_max = fmaxf(fg_max, 1e-12f);
  bg_max = fmaxf(bg_max, 1e-12f);
  for (int pix = tid; pix < HW_; pix += 256) {
    float f = fgd[pix] / fg_max + ((bg_any > 0.f) ? a[pix] / bg_max : 0.f);
    fgd[pix] = (fg_any > 0.f) ? f : 0.f;
  }
  __syncthreads();
}

// ---------------- K1: target distance fields -------------------------------

__global__ void k_target_edt(const float* __restrict__ targets,
                             float* __restrict__ tdt2) {
  __shared__ float msk[HW_], a[HW_], b[HW_], fgd[HW_], red[256];
  const int r = blockIdx.x, tid = threadIdx.x;
  for (int pix = tid; pix < HW_; pix += 256)
    msk[pix] = (targets[r * HW_ + pix] > 0.5f) ? 1.f : 0.f;
  __syncthreads();
  distance_field(msk, a, b, fgd, red, tid);
  for (int pix = tid; pix < HW_; pix += 256)
    tdt2[r * HW_ + pix] = fgd[pix] * fgd[pix];
}

// ---------------- K2: pred distance fields + dist --------------------------

__global__ void k_pred_edt(const float* __restrict__ preds,
                           const float* __restrict__ tdt2,
                           const int* __restrict__ labels,
                           float* __restrict__ dist) {
  __shared__ float msk[HW_], a[HW_], b[HW_], fgd[HW_], red[256];
  const int bid = blockIdx.x, tid = threadIdx.x;
  const int r = bid / C_, c = bid % C_;
  // sigmoid(x) > 0.5  <=>  x > 0
  for (int pix = tid; pix < HW_; pix += 256)
    msk[pix] = (preds[(size_t)bid * HW_ + pix] > 0.f) ? 1.f : 0.f;
  __syncthreads();
  distance_field(msk, a, b, fgd, red, tid);
  const bool lab = (labels[r] == c);
  for (int pix = tid; pix < HW_; pix += 256) {
    float d = fgd[pix] * fgd[pix];
    if (lab) d += tdt2[r * HW_ + pix];
    dist[(size_t)r * P_ + c * HW_ + pix] = d;
  }
}

// ---------------- K3: WMMA triangular cumsum fused with err dot ------------

// Diagonal A-tile of the lower-triangular ones matrix, f16 16x32 layout:
// lane holds row m = lane%16; half h -> K = h + 16*(lane>=16)
__device__ __forceinline__ v16h make_L_tile(int mt, int kc, int lane) {
  v16h am;
  const int mrow = mt * 16 + (lane & 15);
  const int kadd = (lane >> 4) ? 16 : 0;
  #pragma unroll
  for (int h = 0; h < 16; ++h) {
    const int kg = kc * 32 + h + kadd;
    am[h] = (kg <= mrow) ? (_Float16)1.0f : (_Float16)0.0f;
  }
  return am;
}

__global__ void k_wmma_scan(const float* __restrict__ dist,
                            const float* __restrict__ preds,
                            const float* __restrict__ targets,
                            const int* __restrict__ labels,
                            float* __restrict__ partials) {
  __shared__ float chunk[32 * 16];   // dist[kc*32 .. +31, nb .. nb+15]
  __shared__ float red[256];
  const int tid  = threadIdx.x;
  const int nb   = blockIdx.x * 16;
  const int lane = tid & 31;
  const int n = lane & 15, lg = lane >> 4;
  // Wave id as an SGPR so triangular skips become scalar branches and EXEC
  // stays all-ones around every WMMA (ISA requirement).
  const int wid = __builtin_amdgcn_readfirstlane(tid >> 5);
  const int mt0 = wid, mt1 = wid + 8;     // 8 waves x 2 m-tiles = 16 m-tiles
  const int kcmix0 = mt0 >> 1;            // diagonal chunk index for mt0
  const int kcmix1 = mt1 >> 1;            // diagonal chunk index for mt1

  v16h ones;
  #pragma unroll
  for (int h = 0; h < 16; ++h) ones[h] = (_Float16)1.0f;

  v8f acc0 = {};
  v8f acc1 = {};

  for (int kc = 0; kc < 8; ++kc) {
    // stage one 32x16 f32 chunk of dist into LDS
    {
      const int e0 = tid, e1 = tid + 256;
      const float* g0 = &dist[(size_t)(kc * 32 + (e0 >> 4)) * P_ + nb + (e0 & 15)];
      const float* g1 = &dist[(size_t)(kc * 32 + (e1 >> 4)) * P_ + nb + (e1 & 15)];
#if USE_ASYNC_LDS
      async_copy_b32(g0, &chunk[e0]);
      async_copy_b32(g1, &chunk[e1]);
      __builtin_amdgcn_s_wait_asynccnt(0);
#else
      chunk[e0] = *g0;
      chunk[e1] = *g1;
#endif
      if (kc < 7) {  // prefetch next chunk (global_prefetch_b8)
        __builtin_prefetch(&dist[(size_t)((kc + 1) * 32 + (e0 >> 4)) * P_ + nb + (e0 & 15)], 0, 1);
      }
    }
    __syncthreads();

    if (kc <= kcmix1) {   // scalar condition: wave participates this chunk
      // B-tile f16 32x16: lane -> col n = lane%16, half h -> K = h + 16*(lane/16)
      v16h bm;
      const int kbase = lg * 16;
      #pragma unroll
      for (int h = 0; h < 16; ++h)
        bm[h] = (_Float16)chunk[(kbase + h) * 16 + n];

      if (kc <= kcmix0) {
        v16h a0 = (kc < kcmix0) ? ones : make_L_tile(mt0, kc, lane);
        acc0 = __builtin_amdgcn_wmma_f32_16x16x32_f16(false, a0, false, bm,
                                                      (short)0, acc0, false, false);
      }
      v16h a1 = (kc < kcmix1) ? ones : make_L_tile(mt1, kc, lane);
      acc1 = __builtin_amdgcn_wmma_f32_16x16x32_f16(false, a1, false, bm,
                                                    (short)0, acc1, false, false);
    }
    __syncthreads();
  }

  // err-weighted contraction. C/D layout: vgpr v, lane -> M = v + 8*lg, N = n.
  const int pos = nb + n;
  const int c = pos / HW_, hw = pos % HW_;
  float local = 0.f;
  #pragma unroll
  for (int v = 0; v < 8; ++v) {
    {
      const int r = mt0 * 16 + v + 8 * lg;
      const float x = preds[((size_t)r * C_ + c) * HW_ + hw];
      const float p = 1.f / (1.f + expf(-x));
      const float tr = (labels[r] == c) ? targets[r * HW_ + hw] : 0.f;
      const float e = (p - tr) * (p - tr);
      local += e * acc0[v];
    }
    {
      const int r = mt1 * 16 + v + 8 * lg;
      const float x = preds[((size_t)r * C_ + c) * HW_ + hw];
      const float p = 1.f / (1.f + expf(-x));
      const float tr = (labels[r] == c) ? targets[r * HW_ + hw] : 0.f;
      const float e = (p - tr) * (p - tr);
      local += e * acc1[v];
    }
  }

  red[tid] = local;
  __syncthreads();
  for (int s = 128; s > 0; s >>= 1) {
    if (tid < s) red[tid] += red[tid + s];
    __syncthreads();
  }
  if (tid == 0) partials[blockIdx.x] = red[0];
}

// ---------------- K4: deterministic final reduction ------------------------

__global__ void k_reduce(const float* __restrict__ partials,
                         float* __restrict__ out) {
  __shared__ float red[256];
  const int tid = threadIdx.x;
  float s = 0.f;
  for (int i = tid; i < NT_; i += 256) s += partials[i];
  red[tid] = s;
  __syncthreads();
  for (int st = 128; st > 0; st >>= 1) {
    if (tid < st) red[tid] += red[tid + st];
    __syncthreads();
  }
  if (tid == 0) out[0] = red[0] * (1.f / (float)P_);  // mean over (C,H,W)
}

// ---------------- launch ---------------------------------------------------

extern "C" void kernel_launch(void* const* d_in, const int* in_sizes, int n_in,
                              void* d_out, int out_size, void* d_ws, size_t ws_size,
                              hipStream_t stream) {
  (void)in_sizes; (void)n_in; (void)out_size; (void)ws_size;
  const float* preds   = (const float*)d_in[0];   // [R,C,H,W]
  const float* targets = (const float*)d_in[1];   // [R,H,W]
  const int*   labels  = (const int*)d_in[2];     // [R]
  float* out = (float*)d_out;

  float* dist     = (float*)d_ws;                    // R*P floats (~64 MB)
  float* tdt2     = dist + (size_t)R_ * P_;          // R*HW floats
  float* partials = tdt2 + (size_t)R_ * HW_;         // NT_ floats

  k_target_edt<<<R_,       256, 0, stream>>>(targets, tdt2);
  k_pred_edt  <<<R_ * C_,  256, 0, stream>>>(preds, tdt2, labels, dist);
  k_wmma_scan <<<NT_,      256, 0, stream>>>(dist, preds, targets, labels, partials);
  k_reduce    <<<1,        256, 0, stream>>>(partials, out);
}
